// FasterRCNN_17300128268724
// MI455X (gfx1250) — matrix operand order repaired and checked by reading, not compiled, not run on hardware
//
#include <hip/hip_runtime.h>
#include <math.h>

typedef __attribute__((ext_vector_type(16))) __bf16        v16bf;
typedef __attribute__((ext_vector_type(4)))  __bf16        v4bf;
typedef __attribute__((ext_vector_type(8)))  float         v8f;
typedef __attribute__((ext_vector_type(8)))  unsigned int  v8u;
typedef __attribute__((ext_vector_type(4)))  unsigned int  v4u;
typedef __attribute__((ext_vector_type(4)))  float         f32x4;

#define POOL 7
#define NB   2
#define NR   128
#define NC   512
#define NH   38
#define NW   50
#define NROI (NB*NR)          // 256
#define DFLAT (NC*POOL*POOL)  // 25088
#define HID  4096
#define KC   21               // n_classes + 1
#define NBB  (KC*4)           // 84

// ---------------------------------------------------------------------------
// 1) ROI crop + AdaptiveMaxPool2d((7,7)), channel-major flatten
// ---------------------------------------------------------------------------
__global__ __launch_bounds__(256) void roipool_kernel(
    const float* __restrict__ feat,   // [B,C,H,W]
    const int*   __restrict__ rois,   // [B*R,4] (x1,y1,x2,y2) inclusive
    float*       __restrict__ pooled) // [B*R, C*49]
{
    const int n = blockIdx.x;                 // roi index
    const int b = n / NR;
    const int x1 = rois[n*4+0], y1 = rois[n*4+1];
    const int x2 = rois[n*4+2], y2 = rois[n*4+3];
    const int h = y2 - y1 + 1, w = x2 - x1 + 1;

    for (int e = threadIdx.x; e < DFLAT; e += blockDim.x) {
        const int c   = e / 49;
        const int rem = e - c*49;
        const int i   = rem / POOL;
        const int j   = rem - i*POOL;
        const int rs = y1 + (i*h)/POOL;
        const int re = y1 + ((i+1)*h + POOL-1)/POOL;
        const int cs = x1 + (j*w)/POOL;
        const int ce = x1 + ((j+1)*w + POOL-1)/POOL;
        const float* fp = feat + ((size_t)(b*NC + c) * NH) * NW;
        float m = -3.402823466e38f;
        for (int y = rs; y < re; ++y) {
            const float* row = fp + y*NW;
            for (int x = cs; x < ce; ++x) m = fmaxf(m, row[x]);
        }
        pooled[(size_t)n*DFLAT + e] = m;
    }
}

// ---------------------------------------------------------------------------
// 2) GEMM: C[M,N] = A[M,K] @ B[K,N] + bias, fp32 in HBM, bf16 WMMA compute.
//    Workgroup: 256 threads = 8 waves (2 M-groups x 4 N-groups).
//    Tile: 64(M) x 64(N), K-step 32. Each wave: 32x16 -> 2 WMMAs / K-step.
//    LDS rows padded to 40 elems (80 B = 20 banks): conflict-free AND
//    16B-aligned so fragments load as ds_load_b128.
// ---------------------------------------------------------------------------
#define BM 64
#define BN 64
#define KB 32
#define LDA 40
#define LDB 40

template<bool EDGE>
__global__ __launch_bounds__(256) void gemm_bf16_wmma(
    const float* __restrict__ A,     // [M,K] row-major (activations)
    const float* __restrict__ Bw,    // [K,N] row-major (weights)
    const float* __restrict__ bias,  // [N]
    float*       __restrict__ C,     // [M,N] row-major
    int M, int N, int K)
{
    __shared__ __bf16 As[BM * LDA];  // [m][k]
    __shared__ __bf16 Bt[BN * LDB];  // [n][k]  (transposed in LDS)

    const int t    = threadIdx.x;
    const int lane = t & 31;
    const int wv   = t >> 5;
    const int miw  = wv & 1;     // wave's 32-row group within BM
    const int niw  = wv >> 1;    // wave's 16-col group within BN
    const int m0   = blockIdx.y * BM;
    const int n0   = blockIdx.x * BN;

    // A staging: 2 x float4 per thread, row-contiguous in k
    const int rowA0 = t >> 3;              // 0..31 (+32 for second float4)
    const int colA  = (t & 7) << 2;        // 0,4,..,28
    // B staging: one column of 4 k-strided elems per thread (coalesced over n)
    const int nB    = t & 63;              // 0..63
    const int kgB   = (t >> 6) << 2;       // 0,4,8,12 -> k offsets kgB..kgB+3 (x2 halves)

    v8f acc0 = {0.f,0.f,0.f,0.f,0.f,0.f,0.f,0.f};
    v8f acc1 = acc0;

    const int l15 = lane & 15;
    const int kbA = (lane < 16) ? 0 : 8;    // ISA A layout: kbase 0/8, +16 jump at VGPR4
    const int kbB = (lane < 16) ? 0 : 8;    // B layout matches A: kbase 0/8, +16 jump at VGPR4

    const bool bInRange = !EDGE || (n0 + nB) < N;
    const float* Bcol = Bw + (n0 + nB);     // column base (guarded by bInRange)

    f32x4 ra[2], rbv[2];
    // ---- prologue: stage K-tile 0 into registers ----
    #pragma unroll
    for (int i = 0; i < 2; ++i) {
        ra[i] = *(const f32x4*)(A + (size_t)(m0 + rowA0 + i*32)*K + colA);
        #pragma unroll
        for (int j = 0; j < 4; ++j)
            rbv[i][j] = bInRange ? Bcol[(size_t)(kgB + i*16 + j)*N] : 0.f;
    }

    for (int k0 = 0; k0 < K; k0 += KB) {
        __syncthreads();   // previous iteration's LDS reads complete
        // ---- commit staged regs to LDS as bf16 x4 vectors (cvt_pk + b64) ----
        #pragma unroll
        for (int i = 0; i < 2; ++i) {
            v4bf pa;
            pa[0]=(__bf16)ra[i][0];  pa[1]=(__bf16)ra[i][1];
            pa[2]=(__bf16)ra[i][2];  pa[3]=(__bf16)ra[i][3];
            *(v4bf*)&As[(rowA0 + i*32)*LDA + colA] = pa;
            v4bf pb;
            pb[0]=(__bf16)rbv[i][0]; pb[1]=(__bf16)rbv[i][1];
            pb[2]=(__bf16)rbv[i][2]; pb[3]=(__bf16)rbv[i][3];
            *(v4bf*)&Bt[nB*LDB + kgB + i*16] = pb;
        }
        // ---- stage next K-tile into regs (overlaps with WMMAs below) ----
        const int kn = k0 + KB;
        if (kn < K) {
            #pragma unroll
            for (int i = 0; i < 2; ++i) {
                ra[i] = *(const f32x4*)(A + (size_t)(m0 + rowA0 + i*32)*K + kn + colA);
                #pragma unroll
                for (int j = 0; j < 4; ++j)
                    rbv[i][j] = bInRange ? Bcol[(size_t)(kn + kgB + i*16 + j)*N] : 0.f;
            }
            if (kn + KB < K) {   // keep L2 one tile ahead
                __builtin_prefetch(A  + (size_t)(m0 + rowA0)*K + kn + KB + colA, 0, 1);
                __builtin_prefetch(Bw + (size_t)(kn + KB + kgB)*N + n0 + nB, 0, 1);
            }
        }
        __syncthreads();   // LDS tile visible
        // ---- per-lane fragments (ISA 7.12.2 layouts), b128 loads, 2 WMMAs ----
        const __bf16* Ar0 = &As[(miw*32 + l15)*LDA + kbA];
        const __bf16* Ar1 = Ar0 + 16*LDA;
        const __bf16* Br  = &Bt[(niw*16 + l15)*LDB + kbB];
        const v4u a0lo = *(const v4u*)(Ar0);          // K pairs kb+0..7
        const v4u a0hi = *(const v4u*)(Ar0 + 16);     // K pairs kb+16..23
        const v4u a1lo = *(const v4u*)(Ar1);
        const v4u a1hi = *(const v4u*)(Ar1 + 16);
        const v4u blo  = *(const v4u*)(Br);           // K kb+0..7
        const v4u bhi  = *(const v4u*)(Br + 16);      // K kb+16..23
        const v8u a0u = __builtin_shufflevector(a0lo, a0hi, 0,1,2,3,4,5,6,7);
        const v8u a1u = __builtin_shufflevector(a1lo, a1hi, 0,1,2,3,4,5,6,7);
        const v8u bu  = __builtin_shufflevector(blo,  bhi,  0,1,2,3,4,5,6,7);
        const v16bf a0 = __builtin_bit_cast(v16bf, a0u);
        const v16bf a1 = __builtin_bit_cast(v16bf, a1u);
        const v16bf b  = __builtin_bit_cast(v16bf, bu);
        acc0 = __builtin_amdgcn_wmma_f32_16x16x32_bf16(
                   false, a0, false, b, (short)0, acc0, false, false);
        acc1 = __builtin_amdgcn_wmma_f32_16x16x32_bf16(
                   false, a1, false, b, (short)0, acc1, false, false);
    }

    // ---- store: VGPR r -> row r (lanes 0-15) / row r+8 (lanes 16-31) ----
    const int col = n0 + niw*16 + l15;
    if (!EDGE || col < N) {
        const float bsv = bias[col];
        const int   rb0 = m0 + miw*32 + ((lane < 16) ? 0 : 8);
        #pragma unroll
        for (int r = 0; r < 8; ++r) {
            C[(size_t)(rb0 + r     )*N + col] = acc0[r] + bsv;
            C[(size_t)(rb0 + r + 16)*N + col] = acc1[r] + bsv;
        }
    }
}

// ---------------------------------------------------------------------------
// 3) losses: CE (ignore -1, mean) + smooth-L1 on gt-class bbox (sum over pos)
// ---------------------------------------------------------------------------
__global__ __launch_bounds__(256) void loss_kernel(
    const float* __restrict__ score,   // [256,21]
    const float* __restrict__ bbox,    // [256,84]
    const int*   __restrict__ label,   // [256]
    const float* __restrict__ gtb,     // [256,4]
    float*       __restrict__ out)     // [3]
{
    __shared__ float r0[256], r1[256], r2[256];
    const int t   = threadIdx.x;
    const int lab = label[t];
    const bool valid = (lab != -1);
    const int  lbl   = valid ? lab : 0;

    const float* s = score + t*KC;
    float mx = s[0];
    #pragma unroll
    for (int i = 1; i < KC; ++i) mx = fmaxf(mx, s[i]);
    float se = 0.f;
    #pragma unroll
    for (int i = 0; i < KC; ++i) se += __expf(s[i] - mx);
    const float lse = mx + __logf(se);
    const float vf  = valid ? 1.f : 0.f;
    const float nll = (lse - s[lbl]) * vf;

    const float* u   = bbox + t*NBB + lbl*4;
    const float  pos = (lab > 0) ? 1.f : 0.f;
    float bl = 0.f;
    #pragma unroll
    for (int j = 0; j < 4; ++j) {
        const float d = fabsf(gtb[t*4 + j] - u[j]);
        bl += (d < 1.f) ? 0.5f*d*d : (d - 0.5f);
    }
    bl *= pos;

    r0[t] = nll; r1[t] = vf; r2[t] = bl;
    __syncthreads();
    for (int st = 128; st > 0; st >>= 1) {
        if (t < st) { r0[t] += r0[t+st]; r1[t] += r1[t+st]; r2[t] += r2[t+st]; }
        __syncthreads();
    }
    if (t == 0) {
        const float cls = r0[0] / fmaxf(r1[0], 1.f);
        out[0] = cls;
        out[1] = r2[0];
        out[2] = cls + 10.f * r2[0];
    }
}

// ---------------------------------------------------------------------------
extern "C" void kernel_launch(void* const* d_in, const int* in_sizes, int n_in,
                              void* d_out, int out_size, void* d_ws, size_t ws_size,
                              hipStream_t stream) {
    const float* base_feature = (const float*)d_in[0];
    const int*   rois         = (const int*)  d_in[1];
    const int*   gt_label     = (const int*)  d_in[2];
    const float* gt_bbox      = (const float*)d_in[3];
    const float* W1 = (const float*)d_in[4];
    const float* b1 = (const float*)d_in[5];
    const float* W2 = (const float*)d_in[6];
    const float* b2 = (const float*)d_in[7];
    const float* Wb = (const float*)d_in[8];
    const float* bb = (const float*)d_in[9];
    const float* Ws = (const float*)d_in[10];
    const float* bs = (const float*)d_in[11];

    float* pooled = (float*)d_ws;                         // 256 * 25088
    float* x1     = pooled + (size_t)NROI * DFLAT;        // 256 * 4096
    float* x2     = x1     + (size_t)NROI * HID;          // 256 * 4096
    float* cbbox  = x2     + (size_t)NROI * HID;          // 256 * 84
    float* cscore = cbbox  + (size_t)NROI * NBB;          // 256 * 21

    roipool_kernel<<<dim3(NROI), 256, 0, stream>>>(base_feature, rois, pooled);

    gemm_bf16_wmma<false><<<dim3(HID/BN, NROI/BM), 256, 0, stream>>>(
        pooled, W1, b1, x1, NROI, HID, DFLAT);
    gemm_bf16_wmma<false><<<dim3(HID/BN, NROI/BM), 256, 0, stream>>>(
        x1, W2, b2, x2, NROI, HID, HID);
    gemm_bf16_wmma<true><<<dim3((NBB + BN - 1)/BN, NROI/BM), 256, 0, stream>>>(
        x2, Wb, bb, cbbox, NROI, NBB, HID);
    gemm_bf16_wmma<true><<<dim3((KC + BN - 1)/BN, NROI/BM), 256, 0, stream>>>(
        x2, Ws, bs, cscore, NROI, KC, HID);

    loss_kernel<<<1, 256, 0, stream>>>(cscore, cbbox, gt_label, gt_bbox, (float*)d_out);
}
